// STFT_Process_41162966565051
// MI455X (gfx1250) — matrix operand order repaired and checked by reading, not compiled, not run on hardware
//
#include <hip/hip_runtime.h>

// ---------------------------------------------------------------------------
// STFT as WMMA GEMM for MI455X (gfx1250, wave32)
//   C_re[F x (B*frames)] = cosK[F x 4096] * Xframes[4096 x (B*frames)]
//   C_im[F x (B*frames)] = sinK[F x 4096] * Xframes[4096 x (B*frames)]
// f16 inputs, f32 accumulation via v_wmma_f32_16x16x32_f16.
// Global->LDS staging uses CDNA5 async copies (ASYNCcnt) with double buffering.
// k-loop kept rolled (#pragma unroll 1) so accumulators stay register-pinned.
// ---------------------------------------------------------------------------

typedef __attribute__((ext_vector_type(16))) _Float16 v16h;
typedef __attribute__((ext_vector_type(8)))  float    v8f;

#define N_FFT   4096
#define HOP     1024
#define HALF    2048
#define NF      2049                 // freq bins = HALF+1
#define BATCH   8
#define T_IN    524288
#define TPAD    (T_IN + 2 * HALF)    // 528384
#define FRAMES  513

#define TILE_M  128
#define TILE_N  64
#define TILE_K  32
#define NK      (N_FFT / TILE_K)     // 128 k-slabs
#define A_STR   40                   // halfword stride: 80B rows -> 16B aligned, bank-spread
#define B_STR   40

// ---- CDNA5 async global->LDS copy (tracked by ASYNCcnt) --------------------
__device__ __forceinline__ void async_copy_b128(unsigned lds_off, const void* gptr) {
    asm volatile("global_load_async_to_lds_b128 %0, %1, off"
                 :: "v"(lds_off), "v"((unsigned long long)(uintptr_t)gptr)
                 : "memory");
}
__device__ __forceinline__ void wait_async0() {
    asm volatile("s_wait_asynccnt 0x0" ::: "memory");
}
// low 32 bits of a generic LDS pointer == wave-relative LDS byte address
__device__ __forceinline__ unsigned lds_off_of(const void* p) {
    return (unsigned)(uintptr_t)p;
}

// ---- prep kernel 1: reflect-pad + convert input to f16, write zero block ---
__global__ void stft_pad_f16(const float* __restrict__ x, _Float16* __restrict__ xpad,
                             _Float16* __restrict__ zblk) {
    int idx = blockIdx.x * blockDim.x + threadIdx.x;
    if (idx < 32) zblk[idx] = (_Float16)0.0f;       // 64B zero source for OOB async copies
    const int total = BATCH * TPAD;
    if (idx >= total) return;
    int b = idx / TPAD;
    int i = idx - b * TPAD;
    int j = i - HALF;
    if (j < 0) j = -j;                              // reflect (no edge repeat)
    else if (j >= T_IN) j = 2 * T_IN - 2 - j;
    xpad[idx] = (_Float16)x[(size_t)b * T_IN + j];
}

// ---- prep kernel 2: windowed, scaled cos / -sin kernel matrices in f16 -----
__global__ void stft_gen_kernels(_Float16* __restrict__ cosK, _Float16* __restrict__ sinK) {
    int idx = blockIdx.x * blockDim.x + threadIdx.x;
    if (idx >= NF * N_FFT) return;
    int f = idx >> 12;          // / 4096
    int k = idx & (N_FFT - 1);
    const float STEP = 6.283185307179586f / (float)N_FFT;   // 2*pi/N
    // periodic Hann * 1/sqrt(N)
    float w = (0.5f - 0.5f * __cosf(STEP * (float)k)) * (1.0f / 64.0f);
    // exact integer phase reduction: angle = 2*pi*((f*k) mod N)/N
    int m = (f * k) & (N_FFT - 1);
    float s, c;
    __sincosf(STEP * (float)m, &s, &c);
    cosK[idx] = (_Float16)(c * w);
    sinK[idx] = (_Float16)(-s * w);
}

// ---- main GEMM: one WG = 128(freq) x 64(frame) tile of one batch -----------
__global__ __launch_bounds__(256) void stft_wmma_gemm(
    const _Float16* __restrict__ xpad,
    const _Float16* __restrict__ cosK,
    const _Float16* __restrict__ sinK,
    const _Float16* __restrict__ zblk,
    float* __restrict__ out)
{
    // double-buffered tiles: 2 * (10KB cos + 10KB sin + 5KB frames) = 50KB LDS
    __shared__ __align__(16) _Float16 sAc[2][TILE_M * A_STR];
    __shared__ __align__(16) _Float16 sAs[2][TILE_M * A_STR];
    __shared__ __align__(16) _Float16 sB [2][TILE_N * B_STR];

    const int tid  = threadIdx.x;
    const int lane = tid & 31;
    const int wave = tid >> 5;
    const int wm   = (wave & 3) * 32;      // wave's M offset inside tile
    const int wn   = (wave >> 2) * 32;     // wave's N offset inside tile
    const int t0   = blockIdx.x * TILE_N;  // frame base
    const int f0   = blockIdx.y * TILE_M;  // freq base
    const int b    = blockIdx.z;

    // A staging: thread -> (row, 16-wide half of the 32-wide k slab); 32B per matrix
    const int arow  = tid >> 1;
    const int ahalf = (tid & 1) * 16;
    const int af    = min(f0 + arow, NF - 1);           // clamp OOB rows (stores are guarded)
    const _Float16* acsrc = cosK + (size_t)af * N_FFT + ahalf;
    const _Float16* assrc = sinK + (size_t)af * N_FFT + ahalf;

    // B staging: thread -> (frame column, 8-wide chunk of k slab); 16B
    const int bcol   = tid >> 2;
    const int bk     = (tid & 3) * 8;
    const int bt     = t0 + bcol;
    const bool bval  = (bt < FRAMES);
    const _Float16* bsrc = xpad + (size_t)b * TPAD + (size_t)bt * HOP + bk;

    // per-thread LDS destination offsets (buffer 0); buffer 1 adds a fixed delta
    const unsigned ldsAc = lds_off_of(&sAc[0][arow * A_STR + ahalf]);
    const unsigned ldsAs = lds_off_of(&sAs[0][arow * A_STR + ahalf]);
    const unsigned ldsB  = lds_off_of(&sB [0][bcol * B_STR + bk]);
    const unsigned dA = (unsigned)(TILE_M * A_STR * sizeof(_Float16));   // 10240
    const unsigned dB = (unsigned)(TILE_N * B_STR * sizeof(_Float16));   //  5120

    auto stage = [&](int kk, int buf) {
        const unsigned aoff = buf ? dA : 0u;
        const unsigned boff = buf ? dB : 0u;
        const _Float16* gc = acsrc + kk * TILE_K;
        const _Float16* gs = assrc + kk * TILE_K;
        async_copy_b128(ldsAc + aoff,      gc);        // 5 async b128 per thread
        async_copy_b128(ldsAc + aoff + 16, gc + 8);
        async_copy_b128(ldsAs + aoff,      gs);
        async_copy_b128(ldsAs + aoff + 16, gs + 8);
        async_copy_b128(ldsB + boff,
                        bval ? (const void*)(bsrc + kk * TILE_K) : (const void*)zblk);
    };

    v8f accRe[2][2] = {};
    v8f accIm[2][2] = {};

    const int r = lane & 15;
    const int h = lane >> 4;
    // loop-invariant fragment base indices (halfword units)
    const int aBase0 = (wm + 0 * 16 + r) * A_STR + h * 16;
    const int aBase1 = (wm + 1 * 16 + r) * A_STR + h * 16;
    const int bBase0 = (wn + 0 * 16 + r) * B_STR + h * 16;
    const int bBase1 = (wn + 1 * 16 + r) * B_STR + h * 16;

    stage(0, 0);                           // prologue: slab 0 in flight
    #pragma unroll 1                       // keep rolled: accumulators stay register-pinned
    for (int kk = 0; kk < NK; ++kk) {
        const int cur = kk & 1;
        wait_async0();                     // our slab-kk async writes landed in LDS
        __syncthreads();                   // publish; also orders prev compute vs reuse
        if (kk + 1 < NK) stage(kk + 1, cur ^ 1);   // overlap next loads with compute

        const _Float16* __restrict__ pAc = sAc[cur];
        const _Float16* __restrict__ pAs = sAs[cur];
        const _Float16* __restrict__ pB  = sB [cur];

        // fragments in contiguous k order (k = 16h + s), same for A and B
        union Frag { uint4 u[2]; v16h v; };
        Frag fac[2], fas[2], fb[2];
        // A 16x32: lane holds row r, K = h*16 .. h*16+15 contiguous
        fac[0].u[0] = *(const uint4*)&pAc[aBase0];
        fac[0].u[1] = *(const uint4*)&pAc[aBase0 + 8];
        fac[1].u[0] = *(const uint4*)&pAc[aBase1];
        fac[1].u[1] = *(const uint4*)&pAc[aBase1 + 8];
        fas[0].u[0] = *(const uint4*)&pAs[aBase0];
        fas[0].u[1] = *(const uint4*)&pAs[aBase0 + 8];
        fas[1].u[0] = *(const uint4*)&pAs[aBase1];
        fas[1].u[1] = *(const uint4*)&pAs[aBase1 + 8];
        // B 32x16: lane holds col r, K = h*16 .. h*16+15 contiguous (col-major LDS)
        fb[0].u[0] = *(const uint4*)&pB[bBase0];
        fb[0].u[1] = *(const uint4*)&pB[bBase0 + 8];
        fb[1].u[0] = *(const uint4*)&pB[bBase1];
        fb[1].u[1] = *(const uint4*)&pB[bBase1 + 8];

        #pragma unroll
        for (int mi = 0; mi < 2; ++mi)
            #pragma unroll
            for (int ni = 0; ni < 2; ++ni) {
                accRe[mi][ni] = __builtin_amdgcn_wmma_f32_16x16x32_f16(
                    false, fac[mi].v, false, fb[ni].v, (short)0, accRe[mi][ni], false, false);
                accIm[mi][ni] = __builtin_amdgcn_wmma_f32_16x16x32_f16(
                    false, fas[mi].v, false, fb[ni].v, (short)0, accIm[mi][ni], false, false);
            }
    }

    // store: C/D layout -> lane r = N, vgpr row rr -> M = h*8 + rr; pack {re,im}
    #pragma unroll
    for (int mi = 0; mi < 2; ++mi)
        #pragma unroll
        for (int ni = 0; ni < 2; ++ni) {
            int t = t0 + wn + ni * 16 + r;
            if (t >= FRAMES) continue;
            #pragma unroll
            for (int rr = 0; rr < 8; ++rr) {
                int f = f0 + wm + mi * 16 + h * 8 + rr;
                if (f < NF) {
                    size_t o = (((size_t)b * NF + f) * FRAMES + t) * 2;
                    float2 v2 = make_float2(accRe[mi][ni][rr], accIm[mi][ni][rr]);
                    *(float2*)(out + o) = v2;
                }
            }
        }
}

extern "C" void kernel_launch(void* const* d_in, const int* in_sizes, int n_in,
                              void* d_out, int out_size, void* d_ws, size_t ws_size,
                              hipStream_t stream) {
    const float* x  = (const float*)d_in[0];
    float* out      = (float*)d_out;
    char*  ws       = (char*)d_ws;

    // workspace layout (all 16B-aligned by construction):
    //   xpad f16 : 8*528384*2   =  8,454,144 B
    //   cosK f16 : 2049*4096*2  = 16,785,408 B
    //   sinK f16 : 2049*4096*2  = 16,785,408 B
    //   zero blk : 64 B                          (total ~42 MB)
    _Float16* xpad = (_Float16*)ws;
    _Float16* cosK = (_Float16*)(ws + (size_t)BATCH * TPAD * sizeof(_Float16));
    _Float16* sinK = cosK + (size_t)NF * N_FFT;
    _Float16* zblk = sinK + (size_t)NF * N_FFT;

    {
        int n = BATCH * TPAD;
        stft_pad_f16<<<(n + 255) / 256, 256, 0, stream>>>(x, xpad, zblk);
    }
    {
        int n = NF * N_FFT;
        stft_gen_kernels<<<(n + 255) / 256, 256, 0, stream>>>(cosK, sinK);
    }
    dim3 grid((FRAMES + TILE_N - 1) / TILE_N,   // 9
              (NF + TILE_M - 1) / TILE_M,       // 17
              BATCH);                           // 8
    stft_wmma_gemm<<<grid, 256, 0, stream>>>(xpad, cosK, sinK, zblk, out);
}
